// HungarianMatcher_81286551044561
// MI455X (gfx1250) — matrix operand order repaired and checked by reading, not compiled, not run on hardware
//
#include <hip/hip_runtime.h>
#include <hip/hip_bf16.h>
#include <stdint.h>

// ---------------------------------------------------------------------------
// HungarianMatcher for MI455X (gfx1250, wave32)
//   B=16, N=256, M=256, C=4096
//   out = (row_ind int64 [B,N], col_ind int64 [B,N], cost f32 [B,N,M]) flat.
// ---------------------------------------------------------------------------

#define BB   16
#define NN   256
#define MM   256
#define CC   4096
#define INFF 1e30f

typedef __attribute__((ext_vector_type(16))) _Float16     v16h;
typedef __attribute__((ext_vector_type(8)))  float        v8f;
typedef __attribute__((ext_vector_type(4)))  unsigned int v4u;
typedef __attribute__((ext_vector_type(8)))  int          v8i;
typedef __attribute__((ext_vector_type(4)))  int          v4i;

#if defined(__has_builtin)
#  if __has_builtin(__builtin_amdgcn_tensor_load_to_lds) && \
      __has_builtin(__builtin_amdgcn_s_wait_tensorcnt)
#    define HM_HAVE_TDM 1
#  endif
#endif
#ifndef HM_HAVE_TDM
#  define HM_HAVE_TDM 0
#endif

// ---------------------------------------------------------------------------
// Kernel 1: per-(b,n) softmax statistics: max and 1/sum(exp(x-max))
// One 256-thread block per row of 4096 logits; float4 (b128) loads.
// ---------------------------------------------------------------------------
__global__ __launch_bounds__(256)
void hm_softmax_stats(const float* __restrict__ logits, float* __restrict__ stats) {
    const int bn = blockIdx.x;                 // 0 .. B*N-1
    const float4* row4 = (const float4*)(logits + (size_t)bn * CC);  // 1024 float4
    __shared__ float red[256];
    const int t = threadIdx.x;

    float m = -INFF;
    float4 vbuf[4];
#pragma unroll
    for (int it = 0; it < 4; ++it) {
        vbuf[it] = row4[t + 256 * it];
        m = fmaxf(fmaxf(fmaxf(m, vbuf[it].x), fmaxf(vbuf[it].y, vbuf[it].z)), vbuf[it].w);
    }
    red[t] = m; __syncthreads();
    for (int s = 128; s > 0; s >>= 1) {
        if (t < s) red[t] = fmaxf(red[t], red[t + s]);
        __syncthreads();
    }
    const float mx = red[0];
    __syncthreads();

    float sum = 0.f;
#pragma unroll
    for (int it = 0; it < 4; ++it) {
        sum += __expf(vbuf[it].x - mx) + __expf(vbuf[it].y - mx)
             + __expf(vbuf[it].z - mx) + __expf(vbuf[it].w - mx);
    }
    red[t] = sum; __syncthreads();
    for (int s = 128; s > 0; s >>= 1) {
        if (t < s) red[t] += red[t + s];
        __syncthreads();
    }
    if (t == 0) {
        stats[2 * bn]     = mx;
        stats[2 * bn + 1] = 1.0f / red[0];
    }
}

// ---------------------------------------------------------------------------
// Kernel 2: cost tile build via WMMA (unchanged from round 1; lowers to
// v_wmma_f32_16x16x32_f16 with the bezier term fused in the f32 accumulator).
// ---------------------------------------------------------------------------
__global__ __launch_bounds__(32)
void hm_cost_tile(const float* __restrict__ pred_ctrl,     // (B,N,8)
                  const float* __restrict__ logits,        // (B,N,C)
                  const float* __restrict__ gt_ctrl,       // (B,M,8)
                  const long long* __restrict__ gt_labels, // (B,M) int64
                  const float* __restrict__ stats,         // (B*N,2)
                  float* __restrict__ outC) {              // (B,N,M)
    const int mt = blockIdx.x, nt = blockIdx.y, b = blockIdx.z;
    const int lane = threadIdx.x;
    const int n0 = nt * 16, m0 = mt * 16;

    __shared__ int   slbl[16];
    __shared__ float sp[16][8];   // pred rows n0..n0+15
    __shared__ float sg[16][8];   // gt   rows m0..m0+15

    if (lane < 16) slbl[lane] = (int)gt_labels[(size_t)b * MM + m0 + lane];
    {
        const float* pbase = pred_ctrl + ((size_t)b * NN + n0) * 8;
        const float* gbase = gt_ctrl   + ((size_t)b * MM + m0) * 8;
        for (int idx = lane; idx < 128; idx += 32) {
            (&sp[0][0])[idx] = pbase[idx];
            (&sg[0][0])[idx] = gbase[idx];
        }
    }
    __syncthreads();

    // A operand: gathered -prob (f16). 16-bit A 16x32 layout:
    //   lanes 0-15 : row M=lane,    halves 0..7 = K 0..7
    //   lanes 16-31: row M=lane-16, halves 0..7 = K 8..15 ; K16..31 zero.
    const int rowA = lane & 15;
    const int hi   = lane >> 4;
    const int nA   = n0 + rowA;
    const float mx  = stats[2 * ((size_t)b * NN + nA)];
    const float inv = stats[2 * ((size_t)b * NN + nA) + 1];
    const float* lrow = logits + ((size_t)b * NN + nA) * (size_t)CC;
    __builtin_prefetch(lrow, 0, 0);                // global_prefetch_b8

    v16h a;
#pragma unroll
    for (int t = 0; t < 8; ++t) {
        const int j   = hi * 8 + t;                // K index 0..15
        const int cls = slbl[j];
        const float pr = __expf(lrow[cls] - mx) * inv;
        a[t]     = (_Float16)(-pr);                // COST_CLASS = 1
        a[t + 8] = (_Float16)0.0f;
    }

    // B operand: identity embedded in 32x16 (K rows 16..31 zero).
    v16h bI;
#pragma unroll
    for (int h = 0; h < 16; ++h)
        bI[h] = (_Float16)((h < 8 && (lane & 15) == (lane >> 4) * 8 + h) ? 1.0f : 0.0f);

    // C accumulator: 5 * L1 bezier distance, exact f32.
    const int colC  = lane & 15;
    const int rbase = (lane >> 4) * 8;
    v8f c;
#pragma unroll
    for (int vv = 0; vv < 8; ++vv) {
        const int r = rbase + vv;
        float s = 0.f;
#pragma unroll
        for (int k = 0; k < 8; ++k) s += fabsf(sp[r][k] - sg[colC][k]);
        c[vv] = 5.0f * s;                          // COST_BEZIER = 5
    }

    // D = A x I + C (EXEC all-ones here)
    v8f d = __builtin_amdgcn_wmma_f32_16x16x32_f16(
        false, a, false, bI, (short)0, c, false, false);

#pragma unroll
    for (int vv = 0; vv < 8; ++vv) {
        const int r = rbase + vv;
        outC[((size_t)b * NN + n0 + r) * MM + m0 + colC] = d[vv];
    }
}

// ---------------------------------------------------------------------------
// Kernel 3: Jonker-Volgenant shortest augmenting path.
// One wave32 per batch. The batch's full 256KB cost slab is staged into LDS
// (320KB/WGP available) via the Tensor Data Mover, so the serial inner loop
// reads ds_load instead of global_load. Solver state also in LDS.
// ---------------------------------------------------------------------------
__global__ __launch_bounds__(32)
void hm_lsa(const float* __restrict__ cost_all,
            long long* __restrict__ out_row,
            long long* __restrict__ out_col) {
    const int b    = blockIdx.x;
    const int lane = threadIdx.x;

    extern __shared__ float lds_cost[];            // NN*MM floats = 256 KB

    __shared__ float u[NN], v[NN], spc[NN];
    __shared__ int   row4col[NN], col4row[NN], path[NN];
    __shared__ unsigned char SRm[NN], SCm[NN];
    __shared__ float s_minVal;
    __shared__ int   s_i, s_sink;

    const float* gcost = cost_all + (size_t)b * NN * MM;

#if HM_HAVE_TDM
    {
        // Tensor DMA descriptor (ISA 8.3/8.4): 2D tile 256x256 of 4B elements,
        // row stride 256 elements, global -> LDS linear.
        const uint64_t ga = (uint64_t)(uintptr_t)gcost;
        v4u g0;
        g0[0] = 1u;                                         // count=1, user mode
        g0[1] = (unsigned int)(uintptr_t)lds_cost;          // lds_addr (bytes)
        g0[2] = (unsigned int)ga;                           // global_addr[31:0]
        g0[3] = ((unsigned int)(ga >> 32) & 0x01FFFFFFu)    // global_addr[56:32]
              | (2u << 30);                                 // type=2 (image)
        v8i g1;
        g1[0] = 0x00020000;            // data_size=2 (4 bytes), mask=0, flags=0
        g1[1] = (int)(256u << 16);     // tensor_dim0 = 256 (low16 @ bits 63:48)
        g1[2] = (int)(256u << 16);     // tensor_dim0 hi=0, tensor_dim1=256 lo16
        g1[3] = (int)(256u << 16);     // tensor_dim1 hi=0, tile_dim0=256
        g1[4] = 256;                   // tile_dim1 = 256, tile_dim2 = 0
        g1[5] = 256;                   // tensor_dim0_stride = 256 (low 32)
        g1[6] = 0;                     // stride0 hi, tensor_dim1_stride lo = 0
        g1[7] = 0;
        v4i g2 = {0, 0, 0, 0};         // 2D tensor: groups 2/3 unused
        v4i g3 = {0, 0, 0, 0};
#  if __has_include(<hip/amd_detail/amd_gfx1250_TDM.h>)
        v8i gx = {0, 0, 0, 0, 0, 0, 0, 0};
        __builtin_amdgcn_tensor_load_to_lds(g0, g1, g2, g3, gx, 0);
#  else
        __builtin_amdgcn_tensor_load_to_lds(g0, g1, g2, g3, 0);
#  endif
        __builtin_amdgcn_s_wait_tensorcnt(0);
    }
#else
    {
        const float4* src = (const float4*)gcost;
        float4*       dst = (float4*)lds_cost;
        for (int idx = lane; idx < (NN * MM) / 4; idx += 32) dst[idx] = src[idx];
    }
#endif

    for (int t = lane; t < NN; t += 32) {
        u[t] = 0.f; v[t] = 0.f; row4col[t] = -1; col4row[t] = -1;
    }
    __syncthreads();

    for (int cur = 0; cur < NN; ++cur) {
        for (int t = lane; t < NN; t += 32) {
            spc[t] = INFF; path[t] = -1; SRm[t] = 0; SCm[t] = 0;
        }
        if (lane == 0) { s_i = cur; s_minVal = 0.f; s_sink = -1; }
        __syncthreads();

        float minVal = 0.f;
        int   i = cur, sink = -1;

        while (sink < 0) {
            if (lane == 0) SRm[i] = 1;
            const float ui = u[i];
            const float* crow = lds_cost + (size_t)i * MM;   // LDS-resident row

            float lmin = INFF; int lidx = 0;
#pragma unroll
            for (int t = 0; t < 8; ++t) {
                const int j = lane + 32 * t;
                if (!SCm[j]) {
                    const float r = minVal + crow[j] - ui - v[j];
                    if (r < spc[j]) { spc[j] = r; path[j] = i; }
                    const float sj = spc[j];
                    if (sj < lmin) { lmin = sj; lidx = j; }
                }
            }
            // wave32 argmin reduction
            for (int off = 16; off > 0; off >>= 1) {
                const float om = __shfl_down(lmin, off, 32);
                const int   oi = __shfl_down(lidx, off, 32);
                if (om < lmin) { lmin = om; lidx = oi; }
            }
            if (lane == 0) {
                const int jstar = lidx;
                SCm[jstar] = 1;
                s_minVal = lmin;
                const int rc = row4col[jstar];
                if (rc < 0) s_sink = jstar;
                else        s_i = rc;
            }
            __syncthreads();
            minVal = s_minVal; i = s_i; sink = s_sink;
        }

        // dual updates (reference order: before augmentation)
        __syncthreads();
        for (int t = lane; t < NN; t += 32) {
            if (t == cur) {
                u[t] += minVal;
            } else if (SRm[t]) {
                int cr = col4row[t];
                cr = cr < 0 ? 0 : (cr > NN - 1 ? NN - 1 : cr);
                u[t] += minVal - spc[cr];
            }
            if (SCm[t]) v[t] -= minVal - spc[t];
        }
        __syncthreads();

        // augment along predecessor path (serial, lane 0)
        if (lane == 0) {
            int j = sink;
            while (true) {
                const int ii = path[j];
                row4col[j] = ii;
                const int jn = col4row[ii];
                col4row[ii] = j;
                if (ii == cur) break;
                j = jn;
            }
        }
        __syncthreads();
    }

    for (int t = lane; t < NN; t += 32) {
        out_row[(size_t)b * NN + t] = (long long)t;
        out_col[(size_t)b * NN + t] = (long long)col4row[t];
    }
}

// ---------------------------------------------------------------------------
// Host-side launcher
// ---------------------------------------------------------------------------
extern "C" void kernel_launch(void* const* d_in, const int* in_sizes, int n_in,
                              void* d_out, int out_size, void* d_ws, size_t ws_size,
                              hipStream_t stream) {
    const float*     pred_ctrl = (const float*)d_in[0];      // (B,N,4,2)
    const float*     logits    = (const float*)d_in[1];      // (B,N,C)
    const float*     gt_ctrl   = (const float*)d_in[2];      // (B,M,4,2)
    const long long* gt_labels = (const long long*)d_in[3];  // (B,M) int64

    // output packing: row_ind i64 (B,N) | col_ind i64 (B,N) | cost f32 (B,N,M)
    long long* out_row  = (long long*)d_out;
    long long* out_col  = out_row + (size_t)BB * NN;
    float*     out_cost = (float*)(out_col + (size_t)BB * NN);

    float* stats = (float*)d_ws;                             // B*N*2 floats

    hm_softmax_stats<<<BB * NN, 256, 0, stream>>>(logits, stats);

    dim3 tgrid(MM / 16, NN / 16, BB);
    hm_cost_tile<<<tgrid, 32, 0, stream>>>(pred_ctrl, logits, gt_ctrl,
                                           gt_labels, stats, out_cost);

    // 256 KB dynamic LDS: full per-batch cost slab resident on the WGP.
    hm_lsa<<<BB, 32, (size_t)NN * MM * sizeof(float), stream>>>(
        out_cost, out_row, out_col);

    (void)in_sizes; (void)n_in; (void)out_size; (void)ws_size;
}
